// PDECNN3_8839042695231
// MI455X (gfx1250) — matrix operand order repaired and checked by reading, not compiled, not run on hardware
//
#include <hip/hip_runtime.h>

// ---------------- geometry ----------------
#define PPTS   1200       // points per row
#define NCELL  400        // cells
#define NCH    64         // conv channels
#define CHP    72         // padded channel stride in LDS (multiple of 8 -> 16B aligned rows)
#define EPAD   420        // padded extended length (valid 416)
#define NTILE  26         // 16-wide position tiles (26*16 = 416)

// ---------------- LDS layout (bytes, all 16-aligned) ----------------
#define OFF_E     0                      // float[EPAD]            1680
#define OFF_W     1680                   // float[416]             1664 (408 valid)
#define OFF_AN    3344                   // float[400]             1600
#define OFF_IND5  4944                   // float[1200]            4800
#define OFF_MLIM  9744                   // float[1200]            4800
#define OFF_RED   14544                  // float[256+256]         2128 (padded)
#define OFF_BUFA  16672                  // ushort[EPAD*CHP]       60480
#define OFF_BUFB  77152                  // ushort[EPAD*CHP]       60480
#define SMEM_BYTES 137632

// packed-weight offsets (ushort units) in workspace
#define WP_L2 0
#define WP_L3 12288      // 4 mtiles * 6 chunks * 32 lanes * 16
#define WP_L4 24576      // + 12288
#define WP_TOTAL 30720   // + 2*6*32*16

typedef __attribute__((ext_vector_type(16))) __bf16 v16bf;
typedef __attribute__((ext_vector_type(8)))  float  v8f;

union ABfrag {
  v16bf v;
  uint4 q[2];
  unsigned short us[16];
};

__device__ __forceinline__ unsigned short f2bf(float f) {
  unsigned int u = __float_as_uint(f);
  unsigned int r = u + 0x7FFFu + ((u >> 16) & 1u);   // round-to-nearest-even
  return (unsigned short)(r >> 16);
}

__device__ __forceinline__ float sgn(float z) {
  return (z > 0.f) ? 1.f : ((z < 0.f) ? -1.f : 0.f);
}

// Branch-free tanh: one v_exp_f32 + one v_rcp_f32, no EXEC divergence.
// tanh(|x|) = 1 - 2/(exp(2|x|)+1);  exp(2t) = exp2(t * 2*log2(e)).
// |x| large -> exp -> inf -> rcp -> 0 -> result 1.  Sign restored via copysign.
__device__ __forceinline__ float tanh_fast(float x) {
  float t = fabsf(x);
  float e = exp2f(t * 2.88539008177792681472f);       // exp(2t)
  float r = 1.f - 2.f * __builtin_amdgcn_rcpf(e + 1.f);
  return copysignf(r, x);
}

// ---------------------------------------------------------------------------
// Pack conv2/conv3/conv4 weights into bf16 WMMA A-fragment order.
// K ordering: K = t*64 + ci  (t = tap 0..2, ci = input channel).
// A-tile (16x32 bf16): lane m   holds K 0..15 (contiguous, matches B fragment)
//                      lane m+16 holds K 16..31
// frag index = ((mt*6 + kc)*32 + lane)*16 + j  ->  32 contiguous bytes per lane.
// ---------------------------------------------------------------------------
__global__ void pack_weights(const float* __restrict__ w2,
                             const float* __restrict__ w3,
                             const float* __restrict__ w4,
                             unsigned short* __restrict__ wp) {
  int idx = blockIdx.x * 256 + threadIdx.x;
  if (idx >= WP_TOTAL) return;
  const float* w; int base, rows, f;
  if (idx < WP_L3)      { w = w2; f = idx;         base = WP_L2; rows = 64; }
  else if (idx < WP_L4) { w = w3; f = idx - WP_L3; base = WP_L3; rows = 64; }
  else                  { w = w4; f = idx - WP_L4; base = WP_L4; rows = 27; }
  int j    = f & 15;
  int lane = (f >> 4) & 31;
  int kc   = (f >> 9) % 6;
  int mt   = f / 3072;
  int hi = lane >> 4, m = lane & 15;
  int kl = j + 16 * hi;                        // K within 32-chunk
  int K  = kc * 32 + kl;
  int t  = K / 64, ci = K % 64;
  int rowc = mt * 16 + m;
  float val = (rowc < rows) ? w[(rowc * 64 + ci) * 3 + t] : 0.f;
  wp[base + f] = f2bf(val);
}

// ---------------------------------------------------------------------------
// One conv layer as WMMA GEMM: D[64 x 416] = W[64 x 192] * im2col[192 x 416].
// bufIn  : bf16 [pos][ch] (stride CHP)
// !FINAL : tanh + bf16 -> bufOut ; FINAL : raw f32 -> u4f [pos][32]
// All 6 B-fragments are preloaded (12x ds_load_b128 in flight) so the six
// v_wmma_f32_16x16x32_bf16 issue back-to-back on the accumulator chain.
// ---------------------------------------------------------------------------
template <bool FINAL>
__device__ __forceinline__ void conv_wmma(const unsigned short* __restrict__ bufIn,
                                          unsigned short* __restrict__ bufOut,
                                          float* __restrict__ u4f,
                                          const unsigned short* __restrict__ frag,
                                          const float* __restrict__ bias,
                                          int wid, int lane) {
  const int mt   = FINAL ? (wid & 1) : (wid & 3);
  const int nt0  = FINAL ? (wid >> 1) : (wid >> 2);
  const int step = FINAL ? 4 : 2;
  const int hi = lane >> 4, ln = lane & 15;

  ABfrag a[6];
#pragma unroll
  for (int kc = 0; kc < 6; ++kc) {
    const uint4* gp = reinterpret_cast<const uint4*>(frag + (((mt * 6 + kc) * 32 + lane) << 4));
    a[kc].q[0] = gp[0];
    a[kc].q[1] = gp[1];
  }
  v8f cbias;
#pragma unroll
  for (int r = 0; r < 8; ++r) {
    int co = mt * 16 + r + 8 * hi;
    cbias[r] = (!FINAL || co < 27) ? bias[co] : 0.f;
  }

  for (int nt = nt0; nt < NTILE; nt += step) {
    const int x0 = nt << 4;
    ABfrag b[6];
#pragma unroll
    for (int kc = 0; kc < 6; ++kc) {
      const int t   = kc >> 1;
      const int cb  = (kc & 1) << 5;                      // input-channel base 0/32
      const int idx = (x0 + ln + t) * CHP + cb + (hi << 4);
      const uint4* lp = reinterpret_cast<const uint4*>(bufIn + idx);  // 2x ds_load_b128
      b[kc].q[0] = lp[0];
      b[kc].q[1] = lp[1];
    }
    v8f acc = cbias;
#pragma unroll
    for (int kc = 0; kc < 6; ++kc)
      acc = __builtin_amdgcn_wmma_f32_16x16x32_bf16(false, a[kc].v, false, b[kc].v,
                                                    (short)0, acc, false, false);
    if (!FINAL) {
#pragma unroll
      for (int r = 0; r < 8; ++r) {
        int co = mt * 16 + r + 8 * hi;
        bufOut[(x0 + ln) * CHP + co] = f2bf(tanh_fast(acc[r]));
      }
    } else {
#pragma unroll
      for (int r = 0; r < 8; ++r) {
        int ch = mt * 16 + r + 8 * hi;
        u4f[(x0 + ln) * 32 + ch] = acc[r];
      }
    }
  }
}

// ---------------------------------------------------------------------------
// Fully fused PDE-CNN: one workgroup (8 wave32) per batch row.
// ---------------------------------------------------------------------------
__global__ void __launch_bounds__(256)
pde_fused(const float* __restrict__ xt,
          const float* __restrict__ w1, const float* __restrict__ b1,
          const float* __restrict__ b2, const float* __restrict__ b3,
          const float* __restrict__ b4,
          const unsigned short* __restrict__ wpack,
          float* __restrict__ out) {
  extern __shared__ char smem[];
  float*          eL    = reinterpret_cast<float*>(smem + OFF_E);
  float*          wrecL = reinterpret_cast<float*>(smem + OFF_W);
  float*          an    = reinterpret_cast<float*>(smem + OFF_AN);
  float*          ind5L = reinterpret_cast<float*>(smem + OFF_IND5);
  float*          mlim  = reinterpret_cast<float*>(smem + OFF_MLIM);
  float*          red   = reinterpret_cast<float*>(smem + OFF_RED);
  float*          red2  = red + 256;
  unsigned short* bufA  = reinterpret_cast<unsigned short*>(smem + OFF_BUFA);
  unsigned short* bufB  = reinterpret_cast<unsigned short*>(smem + OFF_BUFB);
  // prep-phase aliases (dead before conv buffers are zeroed)
  float* xrow  = reinterpret_cast<float*>(smem + OFF_BUFA);          // 1200 f32
  float* acell = reinterpret_cast<float*>(smem + OFF_BUFA + 4800);   // 400 f32
  float* indT  = reinterpret_cast<float*>(smem + OFF_BUFB);          // 1200 f32

  const int tid  = threadIdx.x;
  const int lane = tid & 31;
  const int wid  = tid >> 5;
  const int row  = blockIdx.x;
  const float* x = xt + (size_t)row * PPTS;

  // ---- 1) load row, cell averages ----
  for (int p = tid; p < PPTS; p += 256) xrow[p] = x[p];
  __syncthreads();
  for (int c = tid; c < NCELL; c += 256)
    acell[c] = (xrow[3 * c] + xrow[3 * c + 1] + xrow[3 * c + 2]) * (1.f / 3.f);
  __syncthreads();

  // ---- 2) mean / unbiased variance over 400 cell averages ----
  {
    float s = 0.f, s2 = 0.f;
    for (int c = tid; c < NCELL; c += 256) { float a = acell[c]; s += a; s2 += a * a; }
    red[tid] = s; red2[tid] = s2;
  }
  __syncthreads();
  for (int off = 128; off > 0; off >>= 1) {
    if (tid < off) { red[tid] += red[tid + off]; red2[tid] += red2[tid + off]; }
    __syncthreads();
  }
  const float mean = red[0] * (1.f / NCELL);
  float var = (red2[0] - (float)NCELL * mean * mean) * (1.f / (NCELL - 1));
  var = fmaxf(var, 0.f);
  const float stdv = sqrtf(var + 1e-8f);
  const float rstd = 1.f / stdv;
  __syncthreads();

  // ---- 3) troubled-cell indicator ----
  for (int p = tid; p < PPTS; p += 256) {
    int c = p / 3;
    float a  = acell[c];
    float d1 = (c < NCELL - 1 ? acell[c + 1] : 0.f) - a;   // Sf boundary -> 0
    float d2 = a - (c > 0 ? acell[c - 1] : 0.f);           // Sb boundary -> 0
    float d3 = a - xrow[p];
    float d4 = -d3;
    float ssum = sgn(d1 + 1e-4f) + sgn(d2 + 1e-4f) + sgn(d3 + 1e-4f) + sgn(d4 + 1e-4f);
    float ind3 = (sgn(fabsf(ssum) - 2.5f) + 1.f) * 0.5f;
    float t1 = sgn(fabsf(d1) - fabsf(d3) + 1e-6f) + sgn(fabsf(d1) - fabsf(d4) + 1e-6f)
             + sgn(fabsf(d2) - fabsf(d3) + 1e-6f) + sgn(fabsf(d2) - fabsf(d4) + 1e-6f);
    float ind4 = (sgn(t1 - 3.5f) + 1.f) * 0.5f;
    indT[p] = 1.f - ind3 * ind4;
  }
  __syncthreads();
  // decaying-max spread
  for (int p = tid; p < PPTS; p += 256) {
    float mx = indT[p];
#pragma unroll
    for (int i = 1; i <= 6; ++i) {
      float cdec = 1.f - (float)i / 7.f;
      int pf = p + 3 * i, pb = p - 3 * i;
      if (pf < PPTS) mx = fmaxf(mx, indT[pf] * cdec);
      if (pb >= 0)   mx = fmaxf(mx, indT[pb] * cdec);
    }
    ind5L[p] = mx;
  }
  // normalized cell averages
  for (int c = tid; c < NCELL; c += 256) an[c] = (acell[c] - mean) * rstd;
  __syncthreads();

  // ---- 4) reflective extension (416) ----
  for (int i = tid; i < EPAD; i += 256) {
    float v;
    if (i < 8)        v = 2.f * an[0] - an[8 - i];
    else if (i < 408) v = an[i - 8];
    else if (i < 416) v = 2.f * an[NCELL - 1] - an[398 - (i - 408)];
    else              v = 0.f;
    eL[i] = v;
  }
  __syncthreads();

  // ---- 5) minlimit + WENO weight ----
  for (int p = tid; p < PPTS; p += 256) {
    int c = p / 3;
    float d1n = (c < NCELL - 1 ? an[c + 1] : 0.f) - an[c];
    float d2n = an[c] - (c > 0 ? an[c - 1] : 0.f);
    mlim[p] = fminf(fminf(fabsf(d1n), fabsf(d2n)), fabsf(d1n + d2n));
  }
  for (int i = tid; i < 408; i += 256) {
    float sw1 = 0.f;
#pragma unroll
    for (int j = 0; j < 8; ++j) { float d = eL[i + j + 1] - eL[i + j]; sw1 += d * d; }
    float sw2 = 0.f;
#pragma unroll
    for (int j = 0; j < 9; ++j) {
      int k = i + j;
      float dd = (k >= 1 && k <= 414) ? (eL[k - 1] - 2.f * eL[k] + eL[k + 1]) : 0.f;
      sw2 += dd * dd;
    }
    wrecL[i] = 1.f / (sw1 + sw2 + 1e-6f);
  }
  __syncthreads();

  // ---- 6) zero both conv buffers (kills prep aliases) ----
  {
    unsigned int* z = reinterpret_cast<unsigned int*>(smem + OFF_BUFA);
    for (int i = tid; i < (2 * EPAD * CHP) / 2; i += 256) z[i] = 0u;
  }
  __syncthreads();

  // ---- 7) conv1 (1 -> 64, VALU) -> bufA bf16 [pos][ch] ----
  {
    const int co = tid & 63;                  // invariant per thread
    const float ww0 = w1[3 * co], ww1 = w1[3 * co + 1], ww2 = w1[3 * co + 2];
    const float bb  = b1[co];
    for (int xp = tid >> 6; xp < 414; xp += 4) {
      float acc = bb + ww0 * eL[xp] + ww1 * eL[xp + 1] + ww2 * eL[xp + 2];
      bufA[xp * CHP + co] = f2bf(tanh_fast(acc));
    }
  }
  __syncthreads();

  // ---- 8) conv2 / conv3 / conv4 via WMMA ----
  conv_wmma<false>(bufA, bufB, nullptr, wpack + WP_L2, b2, wid, lane);
  __syncthreads();
  conv_wmma<false>(bufB, bufA, nullptr, wpack + WP_L3, b3, wid, lane);
  __syncthreads();
  float* u4f = reinterpret_cast<float*>(smem + OFF_BUFB);   // conv2 data dead
  conv_wmma<true>(bufA, nullptr, u4f, wpack + WP_L4, b4, wid, lane);
  __syncthreads();

  // ---- 9) weighted reconstruction -> v[1200] (bufA dead) ----
  float* vF    = reinterpret_cast<float*>(smem + OFF_BUFA);
  float* avgF  = reinterpret_cast<float*>(smem + OFF_BUFA + 4800);
  float* u2F   = reinterpret_cast<float*>(smem + OFF_BUFA + 6400);
  float* avg2F = reinterpret_cast<float*>(smem + OFF_BUFA + 11200);
  for (int p = tid; p < PPTS; p += 256) {
    int c = p / 3, m = p - 3 * c;
    float num = 0.f, den = 0.f;
#pragma unroll
    for (int q = 0; q < 9; ++q) {
      float ww = wrecL[c + q];
      num += ww * u4f[(c + q) * 32 + (m * 9 + q)];
      den += ww;
    }
    vF[p] = num / den;
  }
  __syncthreads();
  // ---- 10) limiter ----
  for (int c = tid; c < NCELL; c += 256)
    avgF[c] = (vF[3 * c] + vF[3 * c + 1] + vF[3 * c + 2]) * (1.f / 3.f);
  __syncthreads();
  for (int p = tid; p < PPTS; p += 256) {
    float u = vF[p] - avgF[p / 3];
    u2F[p] = fminf(fabsf(u), mlim[p]) * sgn(u);
  }
  __syncthreads();
  for (int c = tid; c < NCELL; c += 256)
    avg2F[c] = (u2F[3 * c] + u2F[3 * c + 1] + u2F[3 * c + 2]) * (1.f / 3.f);
  __syncthreads();
  // ---- 11) de-normalize + blend ----
  for (int p = tid; p < PPTS; p += 256) {
    int c = p / 3;
    float u = u2F[p] - avg2F[c] + an[c];
    u = u * stdv + mean;
    float i5 = ind5L[p];
    out[(size_t)row * PPTS + p] = u * i5 + x[p] * (1.f - i5);
  }
}

extern "C" void kernel_launch(void* const* d_in, const int* in_sizes, int n_in,
                              void* d_out, int out_size, void* d_ws, size_t ws_size,
                              hipStream_t stream) {
  const float* xt = (const float*)d_in[0];
  const float* w1 = (const float*)d_in[1];
  const float* b1 = (const float*)d_in[2];
  const float* w2 = (const float*)d_in[3];
  const float* b2 = (const float*)d_in[4];
  const float* w3 = (const float*)d_in[5];
  const float* b3 = (const float*)d_in[6];
  const float* w4 = (const float*)d_in[7];
  const float* b4 = (const float*)d_in[8];
  unsigned short* wp = (unsigned short*)d_ws;

  pack_weights<<<(WP_TOTAL + 255) / 256, 256, 0, stream>>>(w2, w3, w4, wp);

  const int n_rows = in_sizes[0] / PPTS;   // 2048
  pde_fused<<<n_rows, 256, SMEM_BYTES, stream>>>(xt, w1, b1, b2, b3, b4, wp,
                                                 (float*)d_out);
}
